// GCN_LCG_15839839387879
// MI455X (gfx1250) — matrix operand order repaired and checked by reading, not compiled, not run on hardware
//
#include <hip/hip_runtime.h>
#include <cstdint>
#include <cstddef>

#define DIM 128
#define N_ITERS 4

typedef __attribute__((ext_vector_type(16))) _Float16 v16h;
typedef __attribute__((ext_vector_type(8)))  float    v8f;

// ---------------------------------------------------------------------------
// Native CDNA5 fp32 global atomic add (no-return -> STOREcnt), guaranteed
// global_atomic_add_f32 instead of a possible CAS-loop expansion.
// ---------------------------------------------------------------------------
__device__ __forceinline__ void atomic_add_f32(float* p, float v) {
  asm volatile("global_atomic_add_f32 %0, %1, off"
               :
               : "v"(p), "v"(v)
               : "memory");
}

// ---------------------------------------------------------------------------
// WMMA helpers (CDNA5 v_wmma_f32_16x16x32_f16, wave32)
// ---------------------------------------------------------------------------
__device__ __forceinline__ v8f wmma_f16(v16h a, v16h b, v8f c) {
  return __builtin_amdgcn_wmma_f32_16x16x32_f16(
      /*neg_a=*/false, a, /*neg_b=*/false, b,
      /*c_mod=*/(short)0, c, /*reuse_a=*/false, /*reuse_b=*/false);
}

// A fragment 16x32 f16: lane (m = lane&15, kg = (lane>>4)*8) holds
// K = kg+0..7 in halves 0..7 and K = kg+16..23 in halves 8..15.
// p points at rowbase + kstep*32 + kg.
__device__ __forceinline__ v16h a_frag_f32(const float* __restrict__ p) {
  v16h a;
#pragma unroll
  for (int i = 0; i < 8; ++i) a[i] = (_Float16)p[i];
#pragma unroll
  for (int i = 0; i < 8; ++i) a[i + 8] = (_Float16)p[16 + i];
  return a;
}

__device__ __forceinline__ v16h a_frag_f16(const _Float16* __restrict__ p) {
  v16h a;
#pragma unroll
  for (int i = 0; i < 8; ++i) a[i] = p[i];
#pragma unroll
  for (int i = 0; i < 8; ++i) a[i + 8] = p[16 + i];
  return a;
}

// B fragment 32x16 f16 from transposed weights WT[n][k]: lane (n = lane&15)
// holds K = kg+0..7 and kg+16..23, kg = (lane>>4)*8. p = WT + n*ldk + kstep*32 + kg.
__device__ __forceinline__ v16h b_frag(const _Float16* __restrict__ p) {
  v16h b;
#pragma unroll
  for (int i = 0; i < 8; ++i) { b[i] = p[i]; b[i + 8] = p[16 + i]; }
  return b;
}

// ---------------------------------------------------------------------------
// Two-layer MLP: Y = relu(X @ W1 + b1) @ W2 + b2, all 128x128.
// One wave per 16-row tile; 8 column tiles of 16; K=128 = 4 WMMAs each.
// swap_rows folds the literal pos/neg swap (row ^ 1) into the A load.
// ---------------------------------------------------------------------------
__global__ void __launch_bounds__(128)
mlp2_wmma_kernel(const float* __restrict__ X, float* __restrict__ Y, int ntiles,
                 const _Float16* __restrict__ WT1, const float* __restrict__ B1,
                 const _Float16* __restrict__ WT2, const float* __restrict__ B2,
                 int swap_rows)
{
  constexpr int HLD = 136;  // hidden staging stride (halves), padded vs 64 banks
  constexpr int OLD = 132;  // output staging stride (floats), padded
  __shared__ __align__(16) float smem[4][16 * OLD];

  const int wave = threadIdx.x >> 5;
  const int lane = threadIdx.x & 31;
  const int tile = blockIdx.x * 4 + wave;
  if (tile >= ntiles) return;

  const int m  = lane & 15;
  const int kg = (lane >> 4) * 8;   // A sub-K
  const int kb = (lane >> 4) * 16;  // B sub-K
  const int mb = (lane >> 4) * 8;   // C/D row base

  // ---- layer 1 ----
  const int srow = swap_rows ? (m ^ 1) : m;
  const float* arow = X + (size_t)(tile * 16 + srow) * DIM;
  v16h a1[4];
#pragma unroll
  for (int kk = 0; kk < 4; ++kk) a1[kk] = a_frag_f32(arow + kk * 32 + kg);

  _Float16* hbuf = (_Float16*)smem[wave];
#pragma unroll
  for (int nt = 0; nt < 8; ++nt) {
    const float bv = B1[nt * 16 + m];
    v8f c;
#pragma unroll
    for (int r = 0; r < 8; ++r) c[r] = bv;
    const _Float16* wcol = WT1 + (size_t)(nt * 16 + m) * DIM;
#pragma unroll
    for (int kk = 0; kk < 4; ++kk)
      c = wmma_f16(a1[kk], b_frag(wcol + kk * 32 + kg), c);
#pragma unroll
    for (int r = 0; r < 8; ++r) {
      float v = c[r] > 0.f ? c[r] : 0.f;  // ReLU
      hbuf[(mb + r) * HLD + nt * 16 + m] = (_Float16)v;
    }
  }
  asm volatile("" ::: "memory");  // keep f16 stores before f16 reads (alias fence)

  // ---- layer 2 ----
  const _Float16* hrow = hbuf + m * HLD;
  v16h a2[4];
#pragma unroll
  for (int kk = 0; kk < 4; ++kk) a2[kk] = a_frag_f16(hrow + kk * 32 + kg);

  v8f acc[8];
#pragma unroll
  for (int nt = 0; nt < 8; ++nt) {
    const float bv = B2[nt * 16 + m];
    v8f c;
#pragma unroll
    for (int r = 0; r < 8; ++r) c[r] = bv;
    const _Float16* wcol = WT2 + (size_t)(nt * 16 + m) * DIM;
#pragma unroll
    for (int kk = 0; kk < 4; ++kk)
      c = wmma_f16(a2[kk], b_frag(wcol + kk * 32 + kg), c);
    acc[nt] = c;
  }
  asm volatile("" ::: "memory");  // keep f16 reads (a2) before f32 overwrite

  // ---- stage + coalesced store ----
  float* obuf = smem[wave];
#pragma unroll
  for (int nt = 0; nt < 8; ++nt)
#pragma unroll
    for (int r = 0; r < 8; ++r)
      obuf[(mb + r) * OLD + nt * 16 + m] = acc[nt][r];

  float* yrow = Y + (size_t)tile * 16 * DIM;
#pragma unroll
  for (int r = 0; r < 16; ++r) {
    float4 v = *(const float4*)&obuf[r * OLD + lane * 4];
    *(float4*)&yrow[r * DIM + lane * 4] = v;
  }
}

// ---------------------------------------------------------------------------
// Concat update: Y = [X0 | X1 (| X2)] @ W + b, K = 128 * nsrc, no activation.
// ---------------------------------------------------------------------------
__global__ void __launch_bounds__(128)
upd_wmma_kernel(const float* __restrict__ X0, const float* __restrict__ X1,
                const float* __restrict__ X2, int nsrc,
                float* __restrict__ Y, int ntiles,
                const _Float16* __restrict__ WT, int ldk,
                const float* __restrict__ B)
{
  constexpr int OLD = 132;
  __shared__ __align__(16) float smem[4][16 * OLD];

  const int wave = threadIdx.x >> 5;
  const int lane = threadIdx.x & 31;
  const int tile = blockIdx.x * 4 + wave;
  if (tile >= ntiles) return;

  const int m  = lane & 15;
  const int kg = (lane >> 4) * 8;
  const int kb = (lane >> 4) * 16;
  const int mb = (lane >> 4) * 8;

  v8f acc[8];
#pragma unroll
  for (int nt = 0; nt < 8; ++nt) {
    const float bv = B[nt * 16 + m];
#pragma unroll
    for (int r = 0; r < 8; ++r) acc[nt][r] = bv;
  }

  const float* srcs[3] = {X0, X1, X2};
  for (int s = 0; s < nsrc; ++s) {
    const float* arow = srcs[s] + (size_t)(tile * 16 + m) * DIM;
    v16h a[4];
#pragma unroll
    for (int kk = 0; kk < 4; ++kk) a[kk] = a_frag_f32(arow + kk * 32 + kg);
#pragma unroll
    for (int nt = 0; nt < 8; ++nt) {
      const _Float16* wcol = WT + (size_t)(nt * 16 + m) * ldk + s * DIM;
#pragma unroll
      for (int kk = 0; kk < 4; ++kk)
        acc[nt] = wmma_f16(a[kk], b_frag(wcol + kk * 32 + kg), acc[nt]);
    }
  }

  float* obuf = smem[wave];
#pragma unroll
  for (int nt = 0; nt < 8; ++nt)
#pragma unroll
    for (int r = 0; r < 8; ++r)
      obuf[(mb + r) * OLD + nt * 16 + m] = acc[nt][r];

  float* yrow = Y + (size_t)tile * 16 * DIM;
#pragma unroll
  for (int r = 0; r < 16; ++r) {
    float4 v = *(const float4*)&obuf[r * OLD + lane * 4];
    *(float4*)&yrow[r * DIM + lane * 4] = v;
  }
}

// ---------------------------------------------------------------------------
// Graph kernels
// ---------------------------------------------------------------------------
__global__ void degcount_kernel(const int* __restrict__ l_ei,
                                const int* __restrict__ c_ei,
                                float* __restrict__ l_deg,
                                float* __restrict__ c_deg, int E) {
  int i = blockIdx.x * blockDim.x + threadIdx.x;
  if (i < E) {
    atomic_add_f32(&l_deg[l_ei[i]], 1.f);
    atomic_add_f32(&c_deg[c_ei[i]], 1.f);
  }
}

__global__ void edgenorm_kernel(const int* __restrict__ l_ei,
                                const int* __restrict__ c_ei,
                                const float* __restrict__ l_deg,
                                const float* __restrict__ c_deg,
                                float* __restrict__ inv_norm, int E) {
  int i = blockIdx.x * blockDim.x + threadIdx.x;
  if (i < E) inv_norm[i] = rsqrtf(l_deg[l_ei[i]] * c_deg[c_ei[i]]);
}

// One wave per edge: gather both message rows (float4/lane) and atomically
// scatter the degree-normalized messages in both directions using native
// global_atomic_add_f32. Edge scalars are wave-uniform -> readfirstlane so
// the compiler can use scalar loads for them.
__global__ void __launch_bounds__(256)
scatter_kernel(const int* __restrict__ l_ei, const int* __restrict__ c_ei,
               const float* __restrict__ inv_norm,
               const float* __restrict__ l_msg, const float* __restrict__ c_msg,
               float* __restrict__ l2c_aggr, float* __restrict__ c2l_aggr, int E)
{
  int wv = (int)(((long)blockIdx.x * blockDim.x + threadIdx.x) >> 5);
  int lane = threadIdx.x & 31;
  if (wv >= E) return;                       // uniform per wave
  int w = __builtin_amdgcn_readfirstlane(wv);  // provably wave-uniform
  int li = l_ei[w], ci = c_ei[w];
  float s = inv_norm[w];
  const float4 lm = *(const float4*)(l_msg + (size_t)li * DIM + lane * 4);
  const float4 cm = *(const float4*)(c_msg + (size_t)ci * DIM + lane * 4);
  float* pc = l2c_aggr + (size_t)ci * DIM + lane * 4;
  float* pl = c2l_aggr + (size_t)li * DIM + lane * 4;
  atomic_add_f32(pc + 0, s * lm.x); atomic_add_f32(pc + 1, s * lm.y);
  atomic_add_f32(pc + 2, s * lm.z); atomic_add_f32(pc + 3, s * lm.w);
  atomic_add_f32(pl + 0, s * cm.x); atomic_add_f32(pl + 1, s * cm.y);
  atomic_add_f32(pl + 2, s * cm.z); atomic_add_f32(pl + 3, s * cm.w);
}

// ---------------------------------------------------------------------------
// Utility kernels
// ---------------------------------------------------------------------------
__global__ void zero4_kernel(float4* __restrict__ p, long n) {
  long i = (long)blockIdx.x * blockDim.x + threadIdx.x;
  if (i < n) p[i] = make_float4(0.f, 0.f, 0.f, 0.f);
}

__global__ void copy4_kernel(float4* __restrict__ dst,
                             const float4* __restrict__ src, long n) {
  long i = (long)blockIdx.x * blockDim.x + threadIdx.x;
  if (i < n) dst[i] = src[i];
}

// src[K][128] f32  ->  dst[128][K] f16 (transposed, contiguous in K)
__global__ void wtconv_kernel(const float* __restrict__ src,
                              _Float16* __restrict__ dst, int K) {
  int i = blockIdx.x * blockDim.x + threadIdx.x;
  if (i < K * DIM) {
    int n = i / K, k = i % K;
    dst[i] = (_Float16)src[(size_t)k * DIM + n];
  }
}

// ---------------------------------------------------------------------------
// Host launcher
// ---------------------------------------------------------------------------
extern "C" void kernel_launch(void* const* d_in, const int* in_sizes, int n_in,
                              void* d_out, int out_size, void* d_ws, size_t ws_size,
                              hipStream_t stream) {
  (void)n_in; (void)out_size; (void)ws_size;

  const int* l_ei = (const int*)d_in[2];
  const int* c_ei = (const int*)d_in[3];
  const float* l_emb0 = (const float*)d_in[4];
  const float* c_emb0 = (const float*)d_in[5];

  // weights (f32, [K][128]) and biases in dict order
  const float* Wsrc[8] = {
    (const float*)d_in[6],  (const float*)d_in[8],    // l2c W1, W2
    (const float*)d_in[10], (const float*)d_in[12],   // c2l W1, W2
    (const float*)d_in[14], (const float*)d_in[16],   // l2l W1, W2
    (const float*)d_in[18], (const float*)d_in[20] }; // c_upd (K=256), l_upd (K=384)
  const int WK[8] = {128, 128, 128, 128, 128, 128, 256, 384};
  const float* b_l2c1 = (const float*)d_in[7];
  const float* b_l2c2 = (const float*)d_in[9];
  const float* b_c2l1 = (const float*)d_in[11];
  const float* b_c2l2 = (const float*)d_in[13];
  const float* b_l2l1 = (const float*)d_in[15];
  const float* b_l2l2 = (const float*)d_in[17];
  const float* b_cupd = (const float*)d_in[19];
  const float* b_lupd = (const float*)d_in[21];

  const int E = in_sizes[2];
  const int L = in_sizes[4] / DIM;
  const int C = in_sizes[5] / DIM;

  // ---- workspace carve-up ----
  float* ws = (float*)d_ws;
  float* l_deg    = ws; ws += L;
  float* c_deg    = ws; ws += C;
  float* inv_norm = ws; ws += E;
  float* l_msg    = ws; ws += (size_t)L * DIM;
  float* c_msg    = ws; ws += (size_t)C * DIM;
  float* l2l_msg  = ws; ws += (size_t)L * DIM;
  float* l2c_aggr = ws; ws += (size_t)C * DIM;
  float* c2l_aggr = ws; ws += (size_t)L * DIM;
  _Float16* hcur = (_Float16*)ws;
  _Float16* WT[8];
  for (int i = 0; i < 8; ++i) { WT[i] = hcur; hcur += (size_t)WK[i] * DIM; }

  float* out_l = (float*)d_out;                               // [5][L][128]
  float* out_c = out_l + (size_t)(N_ITERS + 1) * L * DIM;     // [5][C][128]

  // ---- one-time prep ----
  {
    long n4 = (long)(L + C) / 4;  // l_deg|c_deg contiguous
    zero4_kernel<<<(unsigned)((n4 + 255) / 256), 256, 0, stream>>>((float4*)l_deg, n4);
  }
  degcount_kernel<<<(E + 255) / 256, 256, 0, stream>>>(l_ei, c_ei, l_deg, c_deg, E);
  edgenorm_kernel<<<(E + 255) / 256, 256, 0, stream>>>(l_ei, c_ei, l_deg, c_deg, inv_norm, E);
  for (int i = 0; i < 8; ++i) {
    int total = WK[i] * DIM;
    wtconv_kernel<<<(total + 255) / 256, 256, 0, stream>>>(Wsrc[i], WT[i], WK[i]);
  }
  {
    long n4 = (long)L * DIM / 4;
    copy4_kernel<<<(unsigned)((n4 + 255) / 256), 256, 0, stream>>>(
        (float4*)out_l, (const float4*)l_emb0, n4);
    n4 = (long)C * DIM / 4;
    copy4_kernel<<<(unsigned)((n4 + 255) / 256), 256, 0, stream>>>(
        (float4*)out_c, (const float4*)c_emb0, n4);
  }

  const int ltiles = L / 16, ctiles = C / 16;
  const unsigned lblk = (unsigned)((ltiles + 3) / 4);
  const unsigned cblk = (unsigned)((ctiles + 3) / 4);

  // ---- message-passing iterations ----
  for (int t = 0; t < N_ITERS; ++t) {
    const float* l_cur = out_l + (size_t)t * L * DIM;
    const float* c_cur = out_c + (size_t)t * C * DIM;
    float* l_nxt = out_l + (size_t)(t + 1) * L * DIM;
    float* c_nxt = out_c + (size_t)(t + 1) * C * DIM;

    mlp2_wmma_kernel<<<lblk, 128, 0, stream>>>(l_cur, l_msg, ltiles,
                                               WT[0], b_l2c1, WT[1], b_l2c2, 0);
    mlp2_wmma_kernel<<<cblk, 128, 0, stream>>>(c_cur, c_msg, ctiles,
                                               WT[2], b_c2l1, WT[3], b_c2l2, 0);
    mlp2_wmma_kernel<<<lblk, 128, 0, stream>>>(l_cur, l2l_msg, ltiles,
                                               WT[4], b_l2l1, WT[5], b_l2l2, 1);

    long n4 = (long)(C + L) * DIM / 4;  // l2c_aggr|c2l_aggr contiguous
    zero4_kernel<<<(unsigned)((n4 + 255) / 256), 256, 0, stream>>>((float4*)l2c_aggr, n4);

    scatter_kernel<<<(unsigned)((E + 7) / 8), 256, 0, stream>>>(
        l_ei, c_ei, inv_norm, l_msg, c_msg, l2c_aggr, c2l_aggr, E);

    upd_wmma_kernel<<<cblk, 128, 0, stream>>>(c_cur, l2c_aggr, nullptr, 2,
                                              c_nxt, ctiles, WT[6], 256, b_cupd);
    upd_wmma_kernel<<<lblk, 128, 0, stream>>>(l_cur, c2l_aggr, l2l_msg, 3,
                                              l_nxt, ltiles, WT[7], 384, b_lupd);
  }
}
